// LocalInferenceModel_41248865911407
// MI455X (gfx1250) — matrix operand order repaired and checked by reading, not compiled, not run on hardware
//
#include <hip/hip_runtime.h>
#include <hip/hip_bf16.h>

#define NEG_INF -10000000.0f

typedef __attribute__((ext_vector_type(16))) __bf16 v16bf;
typedef __attribute__((ext_vector_type(8)))  float  v8f;

// ---------------------------------------------------------------------------
// WMMA helpers (CDNA5 wave32, V_WMMA_F32_16X16X32_BF16)
// ---------------------------------------------------------------------------
__device__ inline v8f wmma_bf16(v16bf a, v16bf b, v8f c) {
    return __builtin_amdgcn_wmma_f32_16x16x32_bf16(
        /*neg_a=*/false, a, /*neg_b=*/false, b,
        /*c_mod=*/(short)0, c, /*reuse_a=*/false, /*reuse_b=*/false);
}

// A-matrix 16x32 bf16 fragment from LDS tile laid out [row][32 k] (k contiguous).
__device__ inline v16bf load_frag_a(const unsigned* lds_u, int row_base, int lane) {
    int m   = lane & 15;
    int khi = lane >> 4;
    union { v16bf v; unsigned u[8]; } f;
    const unsigned* rp = lds_u + (row_base + m) * 16;   // 16 dwords = 32 bf16 per row
#pragma unroll
    for (int i = 0; i < 8; ++i) {
        int kk = (i < 4) ? (2 * i + 8 * khi) : (16 + 2 * (i - 4) + 8 * khi);
        f.u[i] = rp[kk >> 1];
    }
    return f.v;
}

// B-matrix 32x16 bf16 fragment from LDS tile laid out [col n][32 k] (B^T, k contiguous).
__device__ inline v16bf load_frag_b(const unsigned* lds_u, int col_base, int lane) {
    int n  = lane & 15;
    int kh = lane >> 4;
    union { v16bf v; unsigned u[8]; } f;
    const unsigned* rp = lds_u + (col_base + n) * 16;
#pragma unroll
    for (int i = 0; i < 8; ++i) f.u[i] = rp[(i < 4) ? (i + 4 * kh) : (8 + (i - 4) + 4 * kh)];
    return f.v;
}

// Async global -> LDS 16-byte copy (CDNA5, tracked by ASYNCcnt).
__device__ inline void async_copy_b128(unsigned lds_byte_addr, const void* gptr) {
    unsigned long long ga = (unsigned long long)(size_t)gptr;
    asm volatile("global_load_async_to_lds_b128 %0, %1, off"
                 :: "v"(lds_byte_addr), "v"(ga) : "memory");
}
__device__ inline void async_wait0() {
    asm volatile("s_wait_asynccnt 0x0" ::: "memory");
}

// ---------------------------------------------------------------------------
// K0: f32 -> bf16 conversion of p and h
// ---------------------------------------------------------------------------
__global__ __launch_bounds__(256) void cvt_kernel(const float* __restrict__ p,
                                                  const float* __restrict__ h,
                                                  __bf16* __restrict__ pbf,
                                                  __bf16* __restrict__ hbf, int n) {
    int i = blockIdx.x * 256 + threadIdx.x;
    if (i < n) {
        pbf[i] = (__bf16)p[i];
        hbf[i] = (__bf16)h[i];
    }
}

// ---------------------------------------------------------------------------
// K1: E[b,p,h] = p.h^T with mask, 128x128 tile per block, bf16 WMMA.
// Staging via GLOBAL_LOAD_ASYNC_TO_LDS_B128 (no VGPR round trip).
// grid: x = 16 tiles (tr*4+tc), y = batch ; block 256 (8 waves)
// ---------------------------------------------------------------------------
__global__ __launch_bounds__(256) void score_kernel(const __bf16* __restrict__ pbf,
                                                    const __bf16* __restrict__ hbf,
                                                    const int* __restrict__ p_mask,
                                                    const int* __restrict__ h_mask,
                                                    float* __restrict__ E) {
    __shared__ __align__(16) __bf16 As[128 * 32];
    __shared__ __align__(16) __bf16 Bs[128 * 32];
    __shared__ int pm_s[128];
    __shared__ int hm_s[128];

    const int b  = blockIdx.y;
    const int tr = blockIdx.x >> 2;
    const int tc = blockIdx.x & 3;
    const int t    = threadIdx.x;
    const int wave = t >> 5;
    const int lane = t & 31;
    const int pr0 = tr * 128, hr0 = tc * 128;

    if (t < 128)      pm_s[t]       = p_mask[b * 512 + pr0 + t];
    else              hm_s[t - 128] = h_mask[b * 512 + hr0 + (t - 128)];

    v8f acc[8] = {};

    const unsigned* pbu = (const unsigned*)pbf;
    const unsigned* hbu = (const unsigned*)hbf;
    unsigned* asu = (unsigned*)As;
    unsigned* bsu = (unsigned*)Bs;
    const unsigned lds_a = (unsigned)(size_t)(&As[0]);   // low 32 bits = LDS offset
    const unsigned lds_b = (unsigned)(size_t)(&Bs[0]);

    for (int kc = 0; kc < 8; ++kc) {          // K = 256 in chunks of 32
        __syncthreads();                      // prior chunk's ds reads done
        // 512 x b128 per tile, 2 per thread, all 16B aligned
#pragma unroll
        for (int j = 0; j < 2; ++j) {
            int f  = t + 256 * j;             // b128 index 0..511
            int r  = f >> 2;                  // row 0..127
            int dq = f & 3;                   // 16B quarter within 64B row chunk
            async_copy_b128(lds_a + (unsigned)(f * 16),
                            pbu + (size_t)(b * 512 + pr0 + r) * 128 + kc * 16 + dq * 4);
            async_copy_b128(lds_b + (unsigned)(f * 16),
                            hbu + (size_t)(b * 512 + hr0 + r) * 128 + kc * 16 + dq * 4);
        }
        async_wait0();                        // own async copies landed in LDS
        __syncthreads();                      // everyone's copies visible
        v16bf a = load_frag_a(asu, wave * 16, lane);
#pragma unroll
        for (int nt = 0; nt < 8; ++nt) {
            v16bf bb = load_frag_b(bsu, nt * 16, lane);
            acc[nt] = wmma_bf16(a, bb, acc[nt]);
        }
    }

    // masked epilogue: C/D layout m = r + 8*(lane>>4), n = lane&15
    const int hi = lane >> 4, nl = lane & 15;
#pragma unroll
    for (int nt = 0; nt < 8; ++nt) {
#pragma unroll
        for (int r = 0; r < 8; ++r) {
            int lrow = wave * 16 + r + 8 * hi;
            int lcol = nt * 16 + nl;
            float v = acc[nt][r];
            if ((pm_s[lrow] & hm_s[lcol]) == 0) v = NEG_INF;
            E[((size_t)(b * 512 + pr0 + lrow)) * 512 + hr0 + lcol] = v;
        }
    }
}

// ---------------------------------------------------------------------------
// K2: row stats over h axis: rmax/rsum per (b,p).  One block per row.
// ---------------------------------------------------------------------------
__global__ __launch_bounds__(128) void rowstat_kernel(const float* __restrict__ E,
                                                      float* __restrict__ rmax,
                                                      float* __restrict__ rsum) {
    __shared__ float red[128];
    const int row = blockIdx.x;            // b*512 + p
    const int t   = threadIdx.x;
    const float* er = E + (size_t)row * 512;
    float v0 = er[t], v1 = er[t + 128], v2 = er[t + 256], v3 = er[t + 384];
    float mv = fmaxf(fmaxf(v0, v1), fmaxf(v2, v3));
    red[t] = mv;
    __syncthreads();
    for (int s = 64; s > 0; s >>= 1) {
        if (t < s) red[t] = fmaxf(red[t], red[t + s]);
        __syncthreads();
    }
    float M = red[0];
    __syncthreads();
    float sv = __expf(v0 - M) + __expf(v1 - M) + __expf(v2 - M) + __expf(v3 - M);
    red[t] = sv;
    __syncthreads();
    for (int s = 64; s > 0; s >>= 1) {
        if (t < s) red[t] += red[t + s];
        __syncthreads();
    }
    if (t == 0) { rmax[row] = M; rsum[row] = red[0]; }
}

// ---------------------------------------------------------------------------
// K3: column stats over p axis: cmax/csum per (b,h). One block per column.
// ---------------------------------------------------------------------------
__global__ __launch_bounds__(128) void colstat_kernel(const float* __restrict__ E,
                                                      float* __restrict__ cmax,
                                                      float* __restrict__ csum) {
    __shared__ float red[128];
    const int c = blockIdx.x;              // b*512 + h
    const int b = c >> 9, col = c & 511;
    const int t = threadIdx.x;
    const float* ec = E + (size_t)b * 512 * 512 + col;
    float v0 = ec[(size_t)(t)       * 512];
    float v1 = ec[(size_t)(t + 128) * 512];
    float v2 = ec[(size_t)(t + 256) * 512];
    float v3 = ec[(size_t)(t + 384) * 512];
    float mv = fmaxf(fmaxf(v0, v1), fmaxf(v2, v3));
    red[t] = mv;
    __syncthreads();
    for (int s = 64; s > 0; s >>= 1) {
        if (t < s) red[t] = fmaxf(red[t], red[t + s]);
        __syncthreads();
    }
    float M = red[0];
    __syncthreads();
    float sv = __expf(v0 - M) + __expf(v1 - M) + __expf(v2 - M) + __expf(v3 - M);
    red[t] = sv;
    __syncthreads();
    for (int s = 64; s > 0; s >>= 1) {
        if (t < s) red[t] += red[t + s];
        __syncthreads();
    }
    if (t == 0) { cmax[c] = M; csum[c] = red[0]; }
}

// ---------------------------------------------------------------------------
// K4: p_ = p_score @ h  (p_score = column softmax of E), fused m_p epilogue.
// Output tile 128(p) x 128(d).  grid.x = b*8 + mt*2 + nt2
// ---------------------------------------------------------------------------
__global__ __launch_bounds__(256) void mp_kernel(const float* __restrict__ E,
                                                 const __bf16* __restrict__ hbf,
                                                 const float* __restrict__ cmax,
                                                 const float* __restrict__ csum,
                                                 const float* __restrict__ p_in,
                                                 float* __restrict__ out_mp) {
    __shared__ __align__(16) __bf16 As[128 * 32];
    __shared__ __align__(16) __bf16 Bs[128 * 32];
    __shared__ float cm[32], cr[32];

    const int b   = blockIdx.x >> 3;
    const int mt  = (blockIdx.x >> 1) & 3;
    const int nt2 = blockIdx.x & 1;
    const int t = threadIdx.x, wave = t >> 5, lane = t & 31;

    v8f acc[8] = {};
    unsigned* asu = (unsigned*)As;
    unsigned* bsu = (unsigned*)Bs;

    for (int kc = 0; kc < 16; ++kc) {      // K = 512 h positions, chunks of 32
        __syncthreads();
        if (t < 32) {
            cm[t] = cmax[b * 512 + kc * 32 + t];
            cr[t] = 1.0f / csum[b * 512 + kc * 32 + t];
        }
        // prefetch next chunk of E / h while this one is staged
        if (kc < 15) {
            if (t < 128)
                __builtin_prefetch(E + ((size_t)(b * 512 + mt * 128 + t)) * 512 + (kc + 1) * 32, 0, 1);
            else if (t < 160)
                __builtin_prefetch(hbf + ((size_t)(b * 512 + (kc + 1) * 32 + (t - 128))) * 256 + nt2 * 128, 0, 1);
        }
        __syncthreads();
        // A: scores, [128 p][32 h] k-contiguous (coalesced E reads)
#pragma unroll
        for (int j = 0; j < 16; ++j) {
            int f = t + 256 * j;           // 4096 elements
            int r = f >> 5, k = f & 31;
            float ev = E[((size_t)(b * 512 + mt * 128 + r)) * 512 + kc * 32 + k];
            As[r * 32 + k] = (__bf16)(__expf(ev - cm[k]) * cr[k]);
        }
        // B^T: [128 d][32 h], transpose-stage from row-major h bf16
#pragma unroll
        for (int j = 0; j < 16; ++j) {
            int f = t + 256 * j;
            int k = f >> 7, n = f & 127;   // coalesced hbf reads along d
            Bs[n * 32 + k] = hbf[((size_t)(b * 512 + kc * 32 + k)) * 256 + nt2 * 128 + n];
        }
        __syncthreads();
        v16bf a = load_frag_a(asu, wave * 16, lane);
#pragma unroll
        for (int nt = 0; nt < 8; ++nt) {
            v16bf bb = load_frag_b(bsu, nt * 16, lane);
            acc[nt] = wmma_bf16(a, bb, acc[nt]);
        }
    }

    const int hi = lane >> 4, nl = lane & 15;
#pragma unroll
    for (int nt = 0; nt < 8; ++nt) {
#pragma unroll
        for (int r = 0; r < 8; ++r) {
            int prow = mt * 128 + wave * 16 + r + 8 * hi;
            int dcol = nt2 * 128 + nt * 16 + nl;
            float pv = p_in[((size_t)(b * 512 + prow)) * 256 + dcol];
            float pd = acc[nt][r];
            size_t ob = ((size_t)(b * 512 + prow)) * 1024;
            out_mp[ob + dcol]       = pv;
            out_mp[ob + 256 + dcol] = pd;
            out_mp[ob + 512 + dcol] = fabsf(pv - pd);
            out_mp[ob + 768 + dcol] = pv * pd;
        }
    }
}

// ---------------------------------------------------------------------------
// K5: h_ = h_score^T @ p (h_score = row softmax of E), fused m_h epilogue.
// Output tile 128(h) x 128(d).
// ---------------------------------------------------------------------------
__global__ __launch_bounds__(256) void mh_kernel(const float* __restrict__ E,
                                                 const __bf16* __restrict__ pbf,
                                                 const float* __restrict__ rmax,
                                                 const float* __restrict__ rsum,
                                                 const float* __restrict__ h_in,
                                                 float* __restrict__ out_mh) {
    __shared__ __align__(16) __bf16 As[128 * 32];
    __shared__ __align__(16) __bf16 Bs[128 * 32];
    __shared__ float rm[32], rr[32];

    const int b   = blockIdx.x >> 3;
    const int mt  = (blockIdx.x >> 1) & 3;   // h tile
    const int nt2 = blockIdx.x & 1;          // d tile
    const int t = threadIdx.x, wave = t >> 5, lane = t & 31;

    v8f acc[8] = {};
    unsigned* asu = (unsigned*)As;
    unsigned* bsu = (unsigned*)Bs;

    for (int kc = 0; kc < 16; ++kc) {        // K = 512 p positions
        __syncthreads();
        if (t < 32) {
            rm[t] = rmax[b * 512 + kc * 32 + t];
            rr[t] = 1.0f / rsum[b * 512 + kc * 32 + t];
        }
        if (kc < 15) {
            if (t < 32)
                __builtin_prefetch(E + ((size_t)(b * 512 + (kc + 1) * 32 + t)) * 512 + mt * 128, 0, 1);
            else if (t < 64)
                __builtin_prefetch(pbf + ((size_t)(b * 512 + (kc + 1) * 32 + (t - 32))) * 256 + nt2 * 128, 0, 1);
        }
        __syncthreads();
        // A: h_score^T -> [128 h][32 p]; E read transposed (coalesced along h)
#pragma unroll
        for (int j = 0; j < 16; ++j) {
            int f = t + 256 * j;
            int k = f >> 7, r = f & 127;
            float ev = E[((size_t)(b * 512 + kc * 32 + k)) * 512 + mt * 128 + r];
            As[r * 32 + k] = (__bf16)(__expf(ev - rm[k]) * rr[k]);
        }
        // B^T: [128 d][32 p] from row-major p bf16
#pragma unroll
        for (int j = 0; j < 16; ++j) {
            int f = t + 256 * j;
            int k = f >> 7, n = f & 127;
            Bs[n * 32 + k] = pbf[((size_t)(b * 512 + kc * 32 + k)) * 256 + nt2 * 128 + n];
        }
        __syncthreads();
        v16bf a = load_frag_a(asu, wave * 16, lane);
#pragma unroll
        for (int nt = 0; nt < 8; ++nt) {
            v16bf bb = load_frag_b(bsu, nt * 16, lane);
            acc[nt] = wmma_bf16(a, bb, acc[nt]);
        }
    }

    const int hi = lane >> 4, nl = lane & 15;
#pragma unroll
    for (int nt = 0; nt < 8; ++nt) {
#pragma unroll
        for (int r = 0; r < 8; ++r) {
            int hrow = mt * 128 + wave * 16 + r + 8 * hi;
            int dcol = nt2 * 128 + nt * 16 + nl;
            float hv = h_in[((size_t)(b * 512 + hrow)) * 256 + dcol];
            float hd = acc[nt][r];
            size_t ob = ((size_t)(b * 512 + hrow)) * 1024;
            out_mh[ob + dcol]       = hv;
            out_mh[ob + 256 + dcol] = hd;
            out_mh[ob + 512 + dcol] = fabsf(hv - hd);
            out_mh[ob + 768 + dcol] = hv * hd;
        }
    }
}

// ---------------------------------------------------------------------------
// launch
// ---------------------------------------------------------------------------
extern "C" void kernel_launch(void* const* d_in, const int* in_sizes, int n_in,
                              void* d_out, int out_size, void* d_ws, size_t ws_size,
                              hipStream_t stream) {
    const float* p      = (const float*)d_in[0];
    const float* h      = (const float*)d_in[1];
    const int*   p_mask = (const int*)d_in[2];
    const int*   h_mask = (const int*)d_in[3];
    float* out = (float*)d_out;

    // workspace layout
    char* ws = (char*)d_ws;
    float*  E    = (float*)ws;                                  // 64*512*512*4 = 67108864 B
    __bf16* pbf  = (__bf16*)(ws + 67108864);                    // 16777216 B
    __bf16* hbf  = (__bf16*)(ws + 67108864 + 16777216);         // 16777216 B
    float*  rmax = (float*)(ws + 100663296);                    // 64*512 each
    float*  rsum = rmax + 64 * 512;
    float*  cmax = rsum + 64 * 512;
    float*  csum = cmax + 64 * 512;

    const int nElem = 64 * 512 * 256;                           // 8388608 per tensor

    cvt_kernel<<<nElem / 256, 256, 0, stream>>>(p, h, pbf, hbf, nElem);

    dim3 g1(16, 64);
    score_kernel<<<g1, 256, 0, stream>>>(pbf, hbf, p_mask, h_mask, E);

    rowstat_kernel<<<64 * 512, 128, 0, stream>>>(E, rmax, rsum);
    colstat_kernel<<<64 * 512, 128, 0, stream>>>(E, cmax, csum);

    mp_kernel<<<64 * 8, 256, 0, stream>>>(E, hbf, cmax, csum, p, out);
    mh_kernel<<<64 * 8, 256, 0, stream>>>(E, pbf, rmax, rsum, h,
                                          out + (size_t)64 * 512 * 1024);
}
